// Sinkhorn_34660386078953
// MI455X (gfx1250) — matrix operand order repaired and checked by reading, not compiled, not run on hardware
//
#include <hip/hip_runtime.h>

typedef __attribute__((ext_vector_type(16))) _Float16 v16h;
typedef __attribute__((ext_vector_type(8)))  _Float16 v8h;
typedef __attribute__((ext_vector_type(2)))  _Float16 v2h;
typedef __attribute__((ext_vector_type(8)))  float    v8f;

#define NB 256
#define NN 512
#define SINK_ITERS 50

// Fast reciprocal: v_rcp_f32 (~1 ulp), far below the f16-K quantization noise.
__device__ __forceinline__ float fast_rcp(float x) {
    return __builtin_amdgcn_rcpf(x);
}

// ---------------------------------------------------------------------------
// k1: Kh[b,i,j] = (f16) exp(-cost[b,i,j]).  128 MiB f16 -> L2-resident.
// 8 elements per thread: 2x float4 load, 1x 16B f16 store.
// ---------------------------------------------------------------------------
__global__ void sink_expk(const float* __restrict__ cost, _Float16* __restrict__ Kh) {
    size_t idx = ((size_t)blockIdx.x * blockDim.x + threadIdx.x) * 8;
    const float4* c4 = reinterpret_cast<const float4*>(cost + idx);
    float4 c0 = c4[0];
    float4 c1 = c4[1];
    v8h h;
    h[0] = (_Float16)__expf(-c0.x);
    h[1] = (_Float16)__expf(-c0.y);
    h[2] = (_Float16)__expf(-c0.z);
    h[3] = (_Float16)__expf(-c0.w);
    h[4] = (_Float16)__expf(-c1.x);
    h[5] = (_Float16)__expf(-c1.y);
    h[6] = (_Float16)__expf(-c1.z);
    h[7] = (_Float16)__expf(-c1.w);
    *reinterpret_cast<v8h*>(Kh + idx) = h;
}

// ---------------------------------------------------------------------------
// k2: 50 Sinkhorn iterations.  One 1024-thread block (32 waves) per batch.
//   phase v:  y_j = sum_i u_i K[i][j]      (VALU, coalesced over j)
//   phase u:  z_i = sum_j K[i][j] v_j      (v_wmma_f32_16x16x32_f16)
// Emits final log(u), log(v) per batch.
// ---------------------------------------------------------------------------
__global__ __launch_bounds__(1024) void sink_iter(const _Float16* __restrict__ Kh,
                                                  float* __restrict__ lu,
                                                  float* __restrict__ lv) {
    __shared__ float uf[NN];                    // u, f32
    __shared__ float vf[NN];                    // v, f32 (for final log)
    __shared__ __align__(16) _Float16 v16b[NN]; // v, f16 (WMMA B operand)
    __shared__ float ypart[4 * NN];             // column-sum partials

    const int tid = threadIdx.x;
    const int b   = blockIdx.x;
    const _Float16* __restrict__ Kb = Kh + ((size_t)b << 18);  // b * 512*512

    if (tid < NN) uf[tid] = 1.0f / (float)NN;
    __syncthreads();

    // ---- phase-v geometry: 2 columns per thread, i split in quarters ----
    const int jp = (tid & 255) * 2;     // column pair
    const int hq = tid >> 8;            // 0..3 -> i in [hq*128, hq*128+128)
    const _Float16* __restrict__ Kcol = Kb + (size_t)(hq * 128) * NN + jp;
    const float* __restrict__ up = &uf[hq * 128];

    // ---- phase-u geometry: wave w owns rows [16w, 16w+16) ----
    const int lane = tid & 31;
    const int wave = tid >> 5;
    const int s    = lane >> 4;            // half-wave select (ISA A/B layout)
    const int r    = (wave << 4) + (lane & 15);
    const _Float16* __restrict__ Krow = Kb + (size_t)r * NN;

    for (int iter = 0; iter < SINK_ITERS; ++iter) {
        // ---------------- phase v: y = u^T K ----------------
        float a0 = 0.f, a1 = 0.f;
        #pragma unroll 8
        for (int i = 0; i < 128; ++i) {
            float u  = up[i];                                   // wave-uniform LDS broadcast
            v2h  k2  = *reinterpret_cast<const v2h*>(Kcol + (size_t)i * NN);
            a0 += u * (float)k2[0];
            a1 += u * (float)k2[1];
        }
        {
            float2 p; p.x = a0; p.y = a1;
            *reinterpret_cast<float2*>(&ypart[hq * NN + jp]) = p;
        }
        __syncthreads();
        if (tid < NN) {
            float y = (ypart[tid] + ypart[NN + tid]) +
                      (ypart[2 * NN + tid] + ypart[3 * NN + tid]);
            float vv = fast_rcp(y);
            vf[tid]   = vv;
            v16b[tid] = (_Float16)vv;
        }
        __syncthreads();

        // ---------------- phase u: z = K v  via WMMA ----------------
        // D(16x16) += A(16x32) * B(32x16); A = K tile (ISA 16-bit A layout),
        // B[k][n] = v[j0+k] replicated over n -> D rows hold z_i.
        v8f zacc = {0.f, 0.f, 0.f, 0.f, 0.f, 0.f, 0.f, 0.f};
        #pragma unroll 4
        for (int jc = 0; jc < 16; ++jc) {
            const int j0 = jc * 32;
            // A: lane (r, s): halves [j0+8s .. +7] and [j0+16+8s .. +7]
            v8h alo = *reinterpret_cast<const v8h*>(Krow + j0 + 8 * s);
            v8h ahi = *reinterpret_cast<const v8h*>(Krow + j0 + 16 + 8 * s);
            v16h A = __builtin_shufflevector(alo, ahi,
                        0,1,2,3,4,5,6,7,8,9,10,11,12,13,14,15);
            // B: halves v[j0+8s .. +7] and v[j0+16+8s .. +7]
            v8h blo = *reinterpret_cast<const v8h*>(&v16b[j0 + 8 * s]);
            v8h bhi = *reinterpret_cast<const v8h*>(&v16b[j0 + 16 + 8 * s]);
            v16h B = __builtin_shufflevector(blo, bhi,
                        0,1,2,3,4,5,6,7,8,9,10,11,12,13,14,15);
            zacc = __builtin_amdgcn_wmma_f32_16x16x32_f16(
                       false, A, false, B, (short)0, zacc, false, false);
        }
        // C/D layout: VGPR q, lanes 0-15 -> M=q ; lanes 16-31 -> M=8+q.
        // Values are replicated across the 16 N-lanes: lane n==0 writes.
        if ((lane & 15) == 0) {
            #pragma unroll
            for (int q = 0; q < 8; ++q) {
                int row = (wave << 4) + q + 8 * s;
                uf[row] = fast_rcp(zacc[q]);
            }
        }
        __syncthreads();
    }

    if (tid < NN) {
        lu[(size_t)b * NN + tid] = __logf(uf[tid]);
        lv[(size_t)b * NN + tid] = __logf(vf[tid]);
    }
}

// ---------------------------------------------------------------------------
// k3: out[b,i,j] = log u[b,i] + log v[b,j] - cost[b,i,j]   (log K = -cost)
// 4 elements per thread, float4 in/out.  Overwrites all of d_out (incl. the
// Kh scratch region, which is dead by now).
// ---------------------------------------------------------------------------
__global__ void sink_out(const float* __restrict__ cost,
                         const float* __restrict__ lu,
                         const float* __restrict__ lv,
                         float* __restrict__ out) {
    size_t idx = ((size_t)blockIdx.x * blockDim.x + threadIdx.x) * 4;
    size_t b = idx >> 18;
    int    i = (int)((idx >> 9) & (NN - 1));
    int    j = (int)(idx & (NN - 1));
    float  l_u  = lu[b * NN + i];
    float4 lv4  = *reinterpret_cast<const float4*>(lv + b * NN + j);
    float4 c    = *reinterpret_cast<const float4*>(cost + idx);
    float4 o;
    o.x = l_u + lv4.x - c.x;
    o.y = l_u + lv4.y - c.y;
    o.z = l_u + lv4.z - c.z;
    o.w = l_u + lv4.w - c.w;
    *reinterpret_cast<float4*>(out + idx) = o;
}

// ---------------------------------------------------------------------------
extern "C" void kernel_launch(void* const* d_in, const int* in_sizes, int n_in,
                              void* d_out, int out_size, void* d_ws, size_t ws_size,
                              hipStream_t stream) {
    (void)in_sizes; (void)n_in; (void)out_size; (void)ws_size;

    const float* cost = (const float*)d_in[0];
    float*       out  = (float*)d_out;

    // Kh scratch (128 MiB f16) lives in the first half of d_out (256 MiB).
    _Float16* Kh = (_Float16*)d_out;
    // log u / log v in workspace (1 MiB total).
    float* lu = (float*)d_ws;
    float* lv = lu + (size_t)NB * NN;

    const size_t total = (size_t)NB * NN * NN;     // 67,108,864

    sink_expk<<<(unsigned)(total / (256 * 8)), 256, 0, stream>>>(cost, Kh);
    sink_iter<<<NB, 1024, 0, stream>>>(Kh, lu, lv);
    sink_out <<<(unsigned)(total / (256 * 4)), 256, 0, stream>>>(cost, lu, lv, out);
}
